// Model_2534030704774
// MI455X (gfx1250) — compile-verified, matrix operand order checked by reading, not run on hardware
//
#include <hip/hip_runtime.h>

// ---------------- types ----------------
typedef __attribute__((ext_vector_type(16))) __bf16 v16bf;
typedef __attribute__((ext_vector_type(8)))  __bf16 v8bf;
typedef __attribute__((ext_vector_type(8)))  float  v8f;

// ---------------- problem dims ----------------
constexpr int cB  = 128;   // batch
constexpr int cL  = 50;    // history length
constexpr int cT  = 30;    // tokens per title
constexpr int cK1 = 5;     // candidates
constexpr int cD  = 300;   // word dim
constexpr int cDN = 400;   // news dim
constexpr int cH  = 20;    // heads
constexpr int cDU = 420;   // user dim (news 400 + ctr 20)
constexpr int cQV = 200;   // pooling query dim
constexpr int NNEWS = cB * cK1 + cB * cL;   // 7040 (candidates first, then history)

// padded GEMM dims (K multiples of 32, N multiples of 16)
constexpr int KPAD_D  = 320;  // news QKV contraction (300 -> 320)
constexpr int KPAD_DN = 416;  // news pooling contraction (400 -> 416)
constexpr int NPAD_QV = 208;  // pooling output (200 -> 208)
constexpr int KPAD_DU = 448;  // user contraction (420 -> 448)
constexpr int NPAD_DU = 432;  // user QKV output (420 -> 432)

// ---------------- workspace layout (bytes) ----------------
constexpr size_t OFF_NEWSW   = 0;                                  // bf16 [3][400][320]
constexpr size_t SZ_NEWSW    = (size_t)3 * cDN * KPAD_D * 2;       // 768000
constexpr size_t OFF_NEWSA   = OFF_NEWSW + SZ_NEWSW;               // bf16 [208][416]
constexpr size_t SZ_NEWSA    = (size_t)NPAD_QV * KPAD_DN * 2;      // 173056
constexpr size_t OFF_USERW   = OFF_NEWSA + SZ_NEWSA;               // bf16 [3][432][448]
constexpr size_t SZ_USERW    = (size_t)3 * NPAD_DU * KPAD_DU * 2;  // 1161216
constexpr size_t OFF_USERA   = OFF_USERW + SZ_USERW;               // bf16 [208][448]
constexpr size_t SZ_USERA    = (size_t)NPAD_QV * KPAD_DU * 2;      // 186368
constexpr size_t OFF_NEWSVEC = OFF_USERA + SZ_USERA;               // f32 [7040][420]
constexpr size_t SZ_NEWSVEC  = (size_t)NNEWS * cDU * 4;            // 11827200
constexpr size_t OFF_USERVEC = OFF_NEWSVEC + SZ_NEWSVEC;           // f32 [128][420]
constexpr size_t SZ_USERVEC  = (size_t)cB * cDU * 4;               // 215040
constexpr size_t OFF_LOSSB   = OFF_USERVEC + SZ_USERVEC;           // f32 [128]

// ---------------- dynamic LDS layouts ----------------
// news kernel
constexpr int NS_MHX   = 0;                      // bf16: X [32][320] then MH [32][416] = 26624 B
constexpr int NS_Q     = 26624;                  // f32 [32][400] = 51200 B
constexpr int NS_K     = 77824;
constexpr int NS_V     = 129024;
constexpr int NS_TOK   = 180224;                 // int [32]
constexpr int NS_ALPHA = 180352;                 // f32 [32]
constexpr int NS_PART  = 180480;                 // f32 [32][13]
constexpr int NS_RED   = 182144;                 // f32 [4]
constexpr int SMEM_NEWS = 182272;
// user kernel
constexpr int US_MHX   = 0;                      // bf16 [64][448] = 57344 B
constexpr int US_Q     = 57344;                  // bf16 [64][432] = 55296 B
constexpr int US_K     = 112640;
constexpr int US_V     = 167936;
constexpr int US_ALPHA = 223232;                 // f32 [64]
constexpr int US_PART  = 223488;                 // f32 [64][13]
constexpr int US_RED   = 226816;                 // f32 [4]
constexpr int SMEM_USER = 226944;

// ---------------- WMMA helpers ----------------
// CDNA5 16-bit A/B fragment packing (CDNA5 ISA §7.12.2):
// lane m (<16) holds row m, K = {kk..kk+7, kk+16..kk+23}
// lane m (>=16) holds row m-16, K = {kk+8..kk+15, kk+24..kk+31}
// Both halves are 8 contiguous bf16 = one 16B load from a row-major [rows][Kpad] buffer.
__device__ inline v16bf frag_ld(const __bf16* base, int stride, int rc_base, int kk) {
  const int lane = threadIdx.x & 31;
  const int r  = rc_base + (lane & 15);
  const int kq = kk + ((lane < 16) ? 0 : 8);
  const __bf16* p = base + (size_t)r * stride + kq;
  v8bf lo = *(const v8bf*)(p);
  v8bf hi = *(const v8bf*)(p + 16);
  return __builtin_shufflevector(lo, hi, 0,1,2,3,4,5,6,7,8,9,10,11,12,13,14,15);
}

__device__ inline v8f wmma_bf16(v16bf a, v16bf b, v8f c) {
  // D = A x B + C ; f32 accumulate
  return __builtin_amdgcn_wmma_f32_16x16x32_bf16(false, a, false, b, (short)0, c, false, false);
}

// ---------------- weight prep: transpose + pad to bf16 [Np][Kp] ----------------
__global__ void k_transpose_pad(const float* __restrict__ src, __bf16* __restrict__ dst,
                                int K, int N, int Kp, int Np) {
  int idx = blockIdx.x * blockDim.x + threadIdx.x;
  if (idx >= Np * Kp) return;
  int n = idx / Kp, k = idx % Kp;
  float v = (k < K && n < N) ? src[(size_t)k * N + n] : 0.f;
  dst[idx] = (__bf16)v;
}

// ---------------- news encoder: one workgroup per news item ----------------
__global__ void __launch_bounds__(256) k_news_encode(
    const int* __restrict__ cand_tok, const int* __restrict__ hist_tok,
    const float* __restrict__ news_ctr, const float* __restrict__ user_ctr,
    const float* __restrict__ emb,
    const __bf16* __restrict__ WqkvT,          // bf16 [3][400][320]
    const float* __restrict__ bq, const float* __restrict__ bk, const float* __restrict__ bv,
    const __bf16* __restrict__ Aw1T,           // bf16 [208][416]
    const float* __restrict__ ab1, const float* __restrict__ aw2, const float* __restrict__ ab2,
    const float* __restrict__ cw1, const float* __restrict__ cb1,
    const float* __restrict__ cw2, const float* __restrict__ cb2,
    float* __restrict__ newsvec)               // f32 [NNEWS][420]
{
  extern __shared__ char smem[];
  __bf16* sMHX  = (__bf16*)(smem + NS_MHX);
  float*  sQ    = (float*)(smem + NS_Q);
  float*  sK    = (float*)(smem + NS_K);
  float*  sV    = (float*)(smem + NS_V);
  int*    sTok  = (int*)(smem + NS_TOK);
  float*  sAlpha= (float*)(smem + NS_ALPHA);
  float*  sPart = (float*)(smem + NS_PART);
  float*  sRed  = (float*)(smem + NS_RED);

  const int tid = threadIdx.x, lane = tid & 31, wave = tid >> 5;
  const int nidx = blockIdx.x;

  const int* tok;
  float ctr_x;
  if (nidx < cB * cK1) { tok = cand_tok + (size_t)nidx * cT; ctr_x = news_ctr[nidx]; }
  else { int h = nidx - cB * cK1; tok = hist_tok + (size_t)h * cT; ctr_x = user_ctr[h]; }

  if (tid < 32) sTok[tid] = (tid < cT) ? tok[tid] : 0;
  __syncthreads();

  // gather token embeddings -> bf16 X [32][320], zero-padded
  for (int idx = tid; idx < 32 * KPAD_D; idx += 256) {
    int r = idx / KPAD_D, c = idx % KPAD_D;
    float v = (r < cT && c < cD) ? emb[(size_t)sTok[r] * cD + c] : 0.f;
    sMHX[idx] = (__bf16)v;
  }
  __syncthreads();

  // QKV GEMM: 2 M-tiles x 25 N-tiles x 3 matrices = 150 tiles, K-loop 320/32=10
  for (int t = wave; t < 150; t += 8) {
    int mtx = t / 50, rem = t % 50, mt = rem / 25, nt = rem % 25;
    const __bf16* Bt = WqkvT + (size_t)mtx * cDN * KPAD_D;
    __builtin_prefetch(Bt + (size_t)(nt * 16) * KPAD_D, 0, 3);
    v8f acc = {};
    for (int kk = 0; kk < KPAD_D; kk += 32) {
      v16bf a = frag_ld(sMHX, KPAD_D, mt * 16, kk);
      v16bf b = frag_ld(Bt,   KPAD_D, nt * 16, kk);
      acc = wmma_bf16(a, b, acc);
    }
    int col = nt * 16 + (lane & 15);
    int rbase = mt * 16 + ((lane < 16) ? 0 : 8);
    const float* biasp = (mtx == 0) ? bq : (mtx == 1) ? bk : bv;
    float* dstp = (mtx == 0) ? sQ : (mtx == 1) ? sK : sV;
    float bias = biasp[col];
#pragma unroll
    for (int i = 0; i < 8; ++i)
      dstp[(size_t)(rbase + i) * cDN + col] = acc[i] + bias;
  }
  __syncthreads();

  // zero MH region [32][416] bf16 (reuses X space)
  { uint32_t* z = (uint32_t*)sMHX;
    for (int idx = tid; idx < 32 * KPAD_DN / 2; idx += 256) z[idx] = 0u; }
  __syncthreads();

  // per-head attention, d=20, exp-normalized (NRMS style), S=30
  const float scale = 0.223606797749979f;  // 1/sqrt(20)
  for (int p = tid; p < cH * cT; p += 256) {
    int h = p / cT, q = p % cT;
    const float* qp = sQ + (size_t)q * cDN + h * 20;
    float denom = 1e-8f;
    for (int j = 0; j < cT; ++j) {
      const float* kp = sK + (size_t)j * cDN + h * 20;
      float d = 0.f;
      for (int c = 0; c < 20; ++c) d += qp[c] * kp[c];
      denom += expf(d * scale);
    }
    float ctx[20];
    for (int c = 0; c < 20; ++c) ctx[c] = 0.f;
    for (int j = 0; j < cT; ++j) {
      const float* kp = sK + (size_t)j * cDN + h * 20;
      float d = 0.f;
      for (int c = 0; c < 20; ++c) d += qp[c] * kp[c];
      float w = expf(d * scale) / denom;
      const float* vp = sV + (size_t)j * cDN + h * 20;
      for (int c = 0; c < 20; ++c) ctx[c] += w * vp[c];
    }
    __bf16* mh = sMHX + (size_t)q * KPAD_DN + h * 20;
    for (int c = 0; c < 20; ++c) mh[c] = (__bf16)ctx[c];
  }
  __syncthreads();

  // additive pooling: t = tanh(MH @ aw1 + ab1); alpha_raw[r] = sum_j t[r][j]*aw2[j]
  // 2 x 13 tiles; deterministic cross-lane reduction into per-(row,ntile) slots.
  for (int t = wave; t < 26; t += 8) {
    int mt = t / 13, nt = t % 13;
    __builtin_prefetch(Aw1T + (size_t)(nt * 16) * KPAD_DN, 0, 3);
    v8f acc = {};
    for (int kk = 0; kk < KPAD_DN; kk += 32) {
      v16bf a = frag_ld(sMHX, KPAD_DN, mt * 16, kk);
      v16bf b = frag_ld(Aw1T, KPAD_DN, nt * 16, kk);
      acc = wmma_bf16(a, b, acc);
    }
    int col = nt * 16 + (lane & 15);
    float b1v = (col < cQV) ? ab1[col] : 0.f;
    float w2v = (col < cQV) ? aw2[col] : 0.f;
    int rbase = mt * 16 + ((lane < 16) ? 0 : 8);
#pragma unroll
    for (int i = 0; i < 8; ++i) {
      float contrib = tanhf(acc[i] + b1v) * w2v;
      contrib += __shfl_xor(contrib, 1);
      contrib += __shfl_xor(contrib, 2);
      contrib += __shfl_xor(contrib, 4);
      contrib += __shfl_xor(contrib, 8);
      if ((lane & 15) == 0) sPart[(rbase + i) * 13 + nt] = contrib;
    }
  }
  __syncthreads();

  if (tid < 32) {
    float s = 0.f;
    for (int nt = 0; nt < 13; ++nt) s += sPart[tid * 13 + nt];
    sAlpha[tid] = (tid < cT) ? expf(s + ab2[0]) : 0.f;
  }
  __syncthreads();
  if (tid == 0) {
    float s = 0.f;
    for (int r = 0; r < cT; ++r) s += sAlpha[r];
    sRed[0] = 1.f / (s + 1e-8f);
  }
  __syncthreads();

  const float inv = sRed[0];
  for (int c = tid; c < cDN; c += 256) {
    float acc = 0.f;
    for (int r = 0; r < cT; ++r)
      acc += sAlpha[r] * inv * (float)sMHX[(size_t)r * KPAD_DN + c];
    newsvec[(size_t)nidx * cDU + c] = acc;
  }
  // CTR MLP 1->36->20 (LeakyReLU 0.01)
  if (tid < 20) {
    float acc = cb2[tid];
    for (int j = 0; j < 36; ++j) {
      float h = ctr_x * cw1[j] + cb1[j];
      h = (h >= 0.f) ? h : 0.01f * h;
      acc += h * cw2[j * 20 + tid];
    }
    newsvec[(size_t)nidx * cDU + cDN + tid] = acc;
  }
}

// ---------------- user encoder: one workgroup per batch row ----------------
__global__ void __launch_bounds__(256) k_user_encode(
    const float* __restrict__ newsvec, const float* __restrict__ hmask,
    const float* __restrict__ pad_doc,
    const __bf16* __restrict__ WqkvT,          // bf16 [3][432][448]
    const float* __restrict__ bq, const float* __restrict__ bk, const float* __restrict__ bv,
    const __bf16* __restrict__ Aw1T,           // bf16 [208][448]
    const float* __restrict__ ab1, const float* __restrict__ aw2, const float* __restrict__ ab2,
    float* __restrict__ uservec)               // f32 [B][420]
{
  extern __shared__ char smem[];
  __bf16* sMHX  = (__bf16*)(smem + US_MHX);
  __bf16* sQ    = (__bf16*)(smem + US_Q);
  __bf16* sK    = (__bf16*)(smem + US_K);
  __bf16* sV    = (__bf16*)(smem + US_V);
  float*  sAlpha= (float*)(smem + US_ALPHA);
  float*  sPart = (float*)(smem + US_PART);
  float*  sRed  = (float*)(smem + US_RED);

  const int tid = threadIdx.x, lane = tid & 31, wave = tid >> 5;
  const int b = blockIdx.x;

  // X = mask*hist_vec + (1-mask)*pad_doc, bf16 [64][448] zero padded
  for (int idx = tid; idx < 64 * KPAD_DU; idx += 256) {
    int r = idx / KPAD_DU, c = idx % KPAD_DU;
    float v = 0.f;
    if (r < cL && c < cDU) {
      float m  = hmask[b * cL + r];
      float hv = newsvec[((size_t)(cB * cK1) + (size_t)b * cL + r) * cDU + c];
      v = m * hv + (1.f - m) * pad_doc[c];
    }
    sMHX[idx] = (__bf16)v;
  }
  __syncthreads();

  // QKV GEMM: 4 M-tiles x 27 N-tiles x 3 = 324 tiles, K-loop 448/32=14
  for (int t = wave; t < 324; t += 8) {
    int mtx = t / 108, rem = t % 108, mt = rem / 27, nt = rem % 27;
    const __bf16* Bt = WqkvT + (size_t)mtx * NPAD_DU * KPAD_DU;
    __builtin_prefetch(Bt + (size_t)(nt * 16) * KPAD_DU, 0, 3);
    v8f acc = {};
    for (int kk = 0; kk < KPAD_DU; kk += 32) {
      v16bf a = frag_ld(sMHX, KPAD_DU, mt * 16, kk);
      v16bf bm = frag_ld(Bt,  KPAD_DU, nt * 16, kk);
      acc = wmma_bf16(a, bm, acc);
    }
    int col = nt * 16 + (lane & 15);
    int rbase = mt * 16 + ((lane < 16) ? 0 : 8);
    const float* biasp = (mtx == 0) ? bq : (mtx == 1) ? bk : bv;
    __bf16* dstp = (mtx == 0) ? sQ : (mtx == 1) ? sK : sV;
    float bias = (col < cDU) ? biasp[col] : 0.f;
#pragma unroll
    for (int i = 0; i < 8; ++i)
      dstp[(size_t)(rbase + i) * NPAD_DU + col] = (__bf16)(acc[i] + bias);
  }
  __syncthreads();

  { uint32_t* z = (uint32_t*)sMHX;
    for (int idx = tid; idx < 64 * KPAD_DU / 2; idx += 256) z[idx] = 0u; }
  __syncthreads();

  // per-head attention, d=21, S=50
  const float scale = 0.2182178902359924f;  // 1/sqrt(21)
  for (int p = tid; p < cH * cL; p += 256) {
    int h = p / cL, q = p % cL;
    const __bf16* qp = sQ + (size_t)q * NPAD_DU + h * 21;
    float denom = 1e-8f;
    for (int j = 0; j < cL; ++j) {
      const __bf16* kp = sK + (size_t)j * NPAD_DU + h * 21;
      float d = 0.f;
      for (int c = 0; c < 21; ++c) d += (float)qp[c] * (float)kp[c];
      denom += expf(d * scale);
    }
    float ctx[21];
    for (int c = 0; c < 21; ++c) ctx[c] = 0.f;
    for (int j = 0; j < cL; ++j) {
      const __bf16* kp = sK + (size_t)j * NPAD_DU + h * 21;
      float d = 0.f;
      for (int c = 0; c < 21; ++c) d += (float)qp[c] * (float)kp[c];
      float w = expf(d * scale) / denom;
      const __bf16* vp = sV + (size_t)j * NPAD_DU + h * 21;
      for (int c = 0; c < 21; ++c) ctx[c] += w * (float)vp[c];
    }
    __bf16* mh = sMHX + (size_t)q * KPAD_DU + h * 21;
    for (int c = 0; c < 21; ++c) mh[c] = (__bf16)ctx[c];
  }
  __syncthreads();

  // pooling: 4 x 13 tiles
  for (int t = wave; t < 52; t += 8) {
    int mt = t / 13, nt = t % 13;
    __builtin_prefetch(Aw1T + (size_t)(nt * 16) * KPAD_DU, 0, 3);
    v8f acc = {};
    for (int kk = 0; kk < KPAD_DU; kk += 32) {
      v16bf a = frag_ld(sMHX, KPAD_DU, mt * 16, kk);
      v16bf bm = frag_ld(Aw1T, KPAD_DU, nt * 16, kk);
      acc = wmma_bf16(a, bm, acc);
    }
    int col = nt * 16 + (lane & 15);
    float b1v = (col < cQV) ? ab1[col] : 0.f;
    float w2v = (col < cQV) ? aw2[col] : 0.f;
    int rbase = mt * 16 + ((lane < 16) ? 0 : 8);
#pragma unroll
    for (int i = 0; i < 8; ++i) {
      float contrib = tanhf(acc[i] + b1v) * w2v;
      contrib += __shfl_xor(contrib, 1);
      contrib += __shfl_xor(contrib, 2);
      contrib += __shfl_xor(contrib, 4);
      contrib += __shfl_xor(contrib, 8);
      if ((lane & 15) == 0) sPart[(rbase + i) * 13 + nt] = contrib;
    }
  }
  __syncthreads();

  if (tid < 64) {
    float s = 0.f;
    for (int nt = 0; nt < 13; ++nt) s += sPart[tid * 13 + nt];
    sAlpha[tid] = (tid < cL) ? expf(s + ab2[0]) : 0.f;
  }
  __syncthreads();
  if (tid == 0) {
    float s = 0.f;
    for (int r = 0; r < cL; ++r) s += sAlpha[r];
    sRed[0] = 1.f / (s + 1e-8f);
  }
  __syncthreads();

  const float inv = sRed[0];
  for (int c = tid; c < cDU; c += 256) {
    float acc = 0.f;
    for (int r = 0; r < cL; ++r)
      acc += sAlpha[r] * inv * (float)sMHX[(size_t)r * KPAD_DU + c];
    uservec[(size_t)b * cDU + c] = acc;
  }
}

// ---------------- scoring + per-sample log-prob ----------------
__global__ void __launch_bounds__(256) k_score(
    const float* __restrict__ newsvec, const float* __restrict__ uservec,
    const int* __restrict__ label, float* __restrict__ out, float* __restrict__ lossb)
{
  __shared__ float sS[8];
  const int tid = threadIdx.x, lane = tid & 31, wave = tid >> 5;
  const int b = blockIdx.x;
  if (wave < cK1) {
    const float* cv = newsvec + ((size_t)b * cK1 + wave) * cDU;
    const float* uv = uservec + (size_t)b * cDU;
    float acc = 0.f;
    for (int c = lane; c < cDU; c += 32) acc += cv[c] * uv[c];
    acc += __shfl_xor(acc, 16);
    acc += __shfl_xor(acc, 8);
    acc += __shfl_xor(acc, 4);
    acc += __shfl_xor(acc, 2);
    acc += __shfl_xor(acc, 1);
    if (lane == 0) { sS[wave] = acc; out[1 + b * cK1 + wave] = acc; }
  }
  __syncthreads();
  if (tid == 0) {
    float mx = sS[0];
    for (int k = 1; k < cK1; ++k) mx = fmaxf(mx, sS[k]);
    float se = 0.f;
    for (int k = 0; k < cK1; ++k) se += expf(sS[k] - mx);
    float lse = mx + logf(se);
    int lb = label[b];
    lossb[b] = sS[lb] - lse;   // log-prob of the label
  }
}

__global__ void k_loss(const float* __restrict__ lossb, float* __restrict__ out) {
  __shared__ float s[128];
  int tid = threadIdx.x;
  s[tid] = lossb[tid];
  __syncthreads();
  for (int st = 64; st > 0; st >>= 1) {
    if (tid < st) s[tid] += s[tid + st];
    __syncthreads();
  }
  if (tid == 0) out[0] = -s[0] / (float)cB;
}

// ---------------- host launcher ----------------
extern "C" void kernel_launch(void* const* d_in, const int* in_sizes, int n_in,
                              void* d_out, int out_size, void* d_ws, size_t ws_size,
                              hipStream_t stream)
{
  const int*   history   = (const int*)  d_in[0];
  const float* hmask     = (const float*)d_in[1];
  const int*   candidate = (const int*)  d_in[2];
  const int*   label     = (const int*)  d_in[3];
  const float* uctr      = (const float*)d_in[4];
  const float* nctr      = (const float*)d_in[5];
  const float* emb       = (const float*)d_in[6];
  const float* n_wq = (const float*)d_in[7];  const float* n_bq = (const float*)d_in[8];
  const float* n_wk = (const float*)d_in[9];  const float* n_bk = (const float*)d_in[10];
  const float* n_wv = (const float*)d_in[11]; const float* n_bv = (const float*)d_in[12];
  const float* n_aw1= (const float*)d_in[13]; const float* n_ab1= (const float*)d_in[14];
  const float* n_aw2= (const float*)d_in[15]; const float* n_ab2= (const float*)d_in[16];
  const float* c_w1 = (const float*)d_in[17]; const float* c_b1 = (const float*)d_in[18];
  const float* c_w2 = (const float*)d_in[19]; const float* c_b2 = (const float*)d_in[20];
  const float* u_wq = (const float*)d_in[21]; const float* u_bq = (const float*)d_in[22];
  const float* u_wk = (const float*)d_in[23]; const float* u_bk = (const float*)d_in[24];
  const float* u_wv = (const float*)d_in[25]; const float* u_bv = (const float*)d_in[26];
  const float* u_aw1= (const float*)d_in[27]; const float* u_ab1= (const float*)d_in[28];
  const float* u_aw2= (const float*)d_in[29]; const float* u_ab2= (const float*)d_in[30];
  const float* pad_doc = (const float*)d_in[31];

  char* ws = (char*)d_ws;
  __bf16* newsW  = (__bf16*)(ws + OFF_NEWSW);
  __bf16* newsA  = (__bf16*)(ws + OFF_NEWSA);
  __bf16* userW  = (__bf16*)(ws + OFF_USERW);
  __bf16* userA  = (__bf16*)(ws + OFF_USERA);
  float*  newsvec= (float*)(ws + OFF_NEWSVEC);
  float*  uservec= (float*)(ws + OFF_USERVEC);
  float*  lossb  = (float*)(ws + OFF_LOSSB);
  float*  out    = (float*)d_out;

  (void)hipFuncSetAttribute((const void*)k_news_encode,
      hipFuncAttributeMaxDynamicSharedMemorySize, SMEM_NEWS);
  (void)hipFuncSetAttribute((const void*)k_user_encode,
      hipFuncAttributeMaxDynamicSharedMemorySize, SMEM_USER);

  auto tp = [&](const float* src, __bf16* dst, int K, int N, int Kp, int Np) {
    int total = Kp * Np;
    k_transpose_pad<<<(total + 255) / 256, 256, 0, stream>>>(src, dst, K, N, Kp, Np);
  };
  // news QKV weights: (300,400) -> bf16 [400][320]
  tp(n_wq, newsW + (size_t)0 * cDN * KPAD_D, cD, cDN, KPAD_D, cDN);
  tp(n_wk, newsW + (size_t)1 * cDN * KPAD_D, cD, cDN, KPAD_D, cDN);
  tp(n_wv, newsW + (size_t)2 * cDN * KPAD_D, cD, cDN, KPAD_D, cDN);
  // news pooling aw1: (400,200) -> bf16 [208][416]
  tp(n_aw1, newsA, cDN, cQV, KPAD_DN, NPAD_QV);
  // user QKV weights: (420,420) -> bf16 [432][448]
  tp(u_wq, userW + (size_t)0 * NPAD_DU * KPAD_DU, cDU, cDU, KPAD_DU, NPAD_DU);
  tp(u_wk, userW + (size_t)1 * NPAD_DU * KPAD_DU, cDU, cDU, KPAD_DU, NPAD_DU);
  tp(u_wv, userW + (size_t)2 * NPAD_DU * KPAD_DU, cDU, cDU, KPAD_DU, NPAD_DU);
  // user pooling aw1: (420,200) -> bf16 [208][448]
  tp(u_aw1, userA, cDU, cQV, KPAD_DU, NPAD_QV);

  k_news_encode<<<NNEWS, 256, SMEM_NEWS, stream>>>(
      candidate, history, nctr, uctr, emb,
      newsW, n_bq, n_bk, n_bv,
      newsA, n_ab1, n_aw2, n_ab2,
      c_w1, c_b1, c_w2, c_b2, newsvec);

  k_user_encode<<<cB, 256, SMEM_USER, stream>>>(
      newsvec, hmask, pad_doc,
      userW, u_bq, u_bk, u_bv,
      userA, u_ab1, u_aw2, u_ab2, uservec);

  k_score<<<cB, 256, 0, stream>>>(newsvec, uservec, label, out, lossb);
  k_loss<<<1, 128, 0, stream>>>(lossb, out);
}
